// TangentConv_83734682403284
// MI455X (gfx1250) — matrix operand order repaired and checked by reading, not compiled, not run on hardware
//
#include <hip/hip_runtime.h>
#include <hip/hip_bf16.h>

// ---------------------------------------------------------------------------
// Types for CDNA5 WMMA
// ---------------------------------------------------------------------------
typedef __attribute__((ext_vector_type(16))) __bf16 v16bf;
typedef __attribute__((ext_vector_type(8)))  __bf16 v8bf;
typedef __attribute__((ext_vector_type(8)))  float  v8f;

#define WAVES_PER_BLOCK 4

// ---------------------------------------------------------------------------
// Small helper kernels
// ---------------------------------------------------------------------------
__global__ void cvt_f32_to_bf16(const float* __restrict__ in,
                                __bf16* __restrict__ out, int n) {
  int i = blockIdx.x * blockDim.x + threadIdx.x;
  if (i < n) out[i] = (__bf16)in[i];
}

// Repack w (COUT, CPACK, 9) fp32 into native WMMA B fragments, bf16.
// Reduction ordering: r = k*CSRC + c for the source part (size 9*CSRC),
// then r = 9*CSRC + k for the depth column (size 9, if DEPTH), then zero pad.
// Fragment layout: frag-tile ft = s*T + t, element idx = (ft*32 + lane)*16 + j
//   K (reduction) index r = s*32 + 8*(lane>>4) + (j&7) + 16*(j>>3)
//   column (out ch)    o = t*16 + (lane&15)
__global__ void pack_w_kernel(const float* __restrict__ w,
                              __bf16* __restrict__ Bp,
                              int CSRC, int CPACK, int COUT, int SLABS,
                              int DEPTH) {
  int idx = blockIdx.x * blockDim.x + threadIdx.x;
  int T = COUT >> 4;
  int total = SLABS * T * 512;
  if (idx >= total) return;
  int j  = idx & 15;
  int L  = (idx >> 4) & 31;
  int ft = idx >> 9;
  int s  = ft / T;
  int t  = ft - s * T;
  int r  = s * 32 + ((L >> 4) * 8) + (j & 7) + ((j >> 3) * 16);
  int col = t * 16 + (L & 15);
  int Rsrc = CSRC * 9;
  int R = Rsrc + (DEPTH ? 9 : 0);
  float v = 0.f;
  if (r < Rsrc) {
    int k = r / CSRC;
    int c = r - k * CSRC;
    v = w[(col * CPACK + c) * 9 + k];
  } else if (r < R) {
    int k = r - Rsrc;
    v = w[(col * CPACK + CSRC) * 9 + k];
  }
  Bp[idx] = (__bf16)v;
}

// Sum-pool over fan-in 8 (negative index contributes zero), b128 loads
template <int C>
__global__ void pool_kernel(const __bf16* __restrict__ src,
                            const int* __restrict__ pind,
                            __bf16* __restrict__ out, int N) {
  int n = blockIdx.x * blockDim.x + threadIdx.x;
  if (n >= N) return;
  constexpr int V = C / 8;
  v8f acc[V];
#pragma unroll
  for (int v = 0; v < V; ++v) acc[v] = (v8f){0, 0, 0, 0, 0, 0, 0, 0};
#pragma unroll
  for (int j = 0; j < 8; ++j) {
    int s = pind[n * 8 + j];
    if (s >= 0) {
      const v8bf* r = (const v8bf*)(src + (size_t)s * C);
#pragma unroll
      for (int v = 0; v < V; ++v) {
        v8bf x = r[v];
#pragma unroll
        for (int e = 0; e < 8; ++e) acc[v][e] += (float)x[e];
      }
    }
  }
  v8bf* o = (v8bf*)(out + (size_t)n * C);
#pragma unroll
  for (int v = 0; v < V; ++v) {
    v8bf y;
#pragma unroll
    for (int e = 0; e < 8; ++e) y[e] = (__bf16)acc[v][e];
    o[v] = y;
  }
}

// Final 128 -> 1 channel contraction + leaky relu, fp32 output
__global__ void linear_kernel(const __bf16* __restrict__ h,
                              const float* __restrict__ wl,
                              const float* __restrict__ bl,
                              float* __restrict__ out, int N) {
  int n = blockIdx.x * blockDim.x + threadIdx.x;
  if (n >= N) return;
  float acc = bl[0];
  const v8bf* r = (const v8bf*)(h + (size_t)n * 128);
#pragma unroll
  for (int v = 0; v < 16; ++v) {
    v8bf x = r[v];
#pragma unroll
    for (int e = 0; e < 8; ++e) acc += (float)x[e] * wl[v * 8 + e];
  }
  out[n] = acc > 0.f ? acc : 0.1f * acc;
}

// ---------------------------------------------------------------------------
// Tangent-conv as gathered GEMM on WMMA bf16.
// One wave computes a 16-point x COUT tile; reduction swept in 32-wide slabs.
// Reduction ordering (must match pack_w_kernel):
//   r = k*CSRC + c          (source part, 9*CSRC values)
//   r = 9*CSRC + k          (depth column, 9 values, if DEPTH)
// VEC requires CSRC % 16 == 0: then every source slab maps each lane's two
// 8-wide K-runs to contiguous channels of a single gathered row (two 16-byte
// global loads), and the depth part is exactly one extra slab built from LDS.
// Non-VEC path (only CSRC=3 layer): slab staged in LDS column-per-lane, so
// all data-dependent conditions are hoisted out of the inner copy loop.
// ---------------------------------------------------------------------------
template <int CSRC, bool DEPTH, int COUT, bool VEC>
__global__ __launch_bounds__(WAVES_PER_BLOCK * 32)
void conv_kernel(const __bf16* __restrict__ src,
                 const int* __restrict__ ind,
                 const float* __restrict__ depth,
                 const __bf16* __restrict__ Bp,
                 const float* __restrict__ bias,
                 __bf16* __restrict__ out, int N) {
  constexpr int KK = 9;
  constexpr int RSRC = CSRC * KK;
  constexpr int R = RSRC + (DEPTH ? KK : 0);
  constexpr int SLABS = (R + 31) / 32;
  constexpr int FULL = VEC ? (RSRC / 32) : 0;
  constexpr int T = COUT / 16;
  static_assert(!VEC || (RSRC % 32 == 0 && CSRC % 16 == 0), "VEC layout");

  __shared__ int    indS[WAVES_PER_BLOCK][16 * KK];
  __shared__ float  depS[WAVES_PER_BLOCK][16 * KK];
  __shared__ __bf16 aS[WAVES_PER_BLOCK][16][32];  // used by non-VEC path

  const int lane = threadIdx.x & 31;
  const int wave = threadIdx.x >> 5;
  const int p0   = (blockIdx.x * WAVES_PER_BLOCK + wave) * 16;
  const int half = lane >> 4;
  const int sub  = lane & 15;

  // Stage this wave's 16x9 gather indices (+depth) into LDS.
  for (int t = lane; t < 16 * KK; t += 32) {
    const int p = p0 + t / KK;
    int idx = 0;
    float d = 0.f;
    if (p < N) {
      idx = ind[p * KK + t % KK];
      if (DEPTH) d = depth[p * KK + t % KK];
    }
    indS[wave][t] = idx;
    if (DEPTH) depS[wave][t] = d;
  }
  __syncthreads();

  v8f acc[T];
#pragma unroll
  for (int t = 0; t < T; ++t) acc[t] = (v8f){0, 0, 0, 0, 0, 0, 0, 0};

  auto mma = [&](int s, v16bf a) {
#pragma unroll
    for (int t = 0; t < T; ++t) {
      v16bf b = *(const v16bf*)(Bp + ((size_t)(s * T + t) * 32 + lane) * 16);
      acc[t] = __builtin_amdgcn_wmma_f32_16x16x32_bf16(
          false, a, false, b, (short)0, acc[t], false, false);
    }
  };

  if (VEC) {
    // Hoist this lane's 9 gather row indices into registers.
    int rowIdx[KK];
#pragma unroll
    for (int k = 0; k < KK; ++k) rowIdx[k] = indS[wave][sub * KK + k];

#pragma unroll
    for (int s = 0; s < FULL; ++s) {
      const int k  = (s * 32) / CSRC;               // compile-time
      const int cb = (s * 32) % CSRC + 8 * half;
      const __bf16* q = src + (size_t)rowIdx[k] * CSRC + cb;
      union { v16bf v; v8bf h[2]; } au;
      au.h[0] = *(const v8bf*)(q);        // K-run {cb .. cb+7}
      au.h[1] = *(const v8bf*)(q + 16);   // K-run {cb+16 .. cb+23}
      mma(s, au.v);
    }
    if (DEPTH) {
      // One extra slab holding the 9 depth values (branchless build).
      v16bf a;
#pragma unroll
      for (int j = 0; j < 16; ++j) {
        const int kk2 = 8 * half + j + ((j >> 3) << 3);  // K within slab
        const int kcl = kk2 < KK ? kk2 : 0;
        const float dv = depS[wave][sub * KK + kcl];
        a[j] = (__bf16)(kk2 < KK ? dv : 0.f);
      }
      mma(FULL, a);
    }
  } else {
#pragma unroll
    for (int s = 0; s < SLABS; ++s) {
      // Column-per-lane staging: lane owns reduction index r = s*32+lane for
      // all 16 rows, so all conditions are uniform per lane per slab.
      const int r = s * 32 + lane;
      const bool isSrc = r < RSRC;
      const bool isDep = DEPTH && !isSrc && (r < R);
      if (isSrc) {
        const int k = r / CSRC;
        const int c = r - k * CSRC;
#pragma unroll
        for (int p = 0; p < 16; ++p)
          aS[wave][p][lane] =
              src[(size_t)indS[wave][p * KK + k] * CSRC + c];
      } else if (isDep) {
        const int k = r - RSRC;
#pragma unroll
        for (int p = 0; p < 16; ++p)
          aS[wave][p][lane] = (__bf16)depS[wave][p * KK + k];
      } else {
#pragma unroll
        for (int p = 0; p < 16; ++p) aS[wave][p][lane] = (__bf16)0.f;
      }
      __builtin_amdgcn_wave_barrier();  // LDS ops are in-order per wave
      union { v16bf v; v8bf h[2]; } au;
      au.h[0] = *(const v8bf*)(&aS[wave][sub][8 * half]);
      au.h[1] = *(const v8bf*)(&aS[wave][sub][16 + 8 * half]);
      __builtin_amdgcn_wave_barrier();
      mma(s, au.v);
    }
  }

  // Epilogue: bias + leaky relu, store per C/D layout
  // (VGPR i <-> row i + 8*(lane>>4); col = lane&15)
#pragma unroll
  for (int t = 0; t < T; ++t) {
    const int ch = t * 16 + sub;
    const float bv = bias[ch];
#pragma unroll
    for (int i = 0; i < 8; ++i) {
      const int p = p0 + 8 * half + i;
      if (p < N) {
        float v = acc[t][i] + bv;
        v = v > 0.f ? v : 0.1f * v;
        out[(size_t)p * COUT + ch] = (__bf16)v;
      }
    }
  }
}

// ---------------------------------------------------------------------------
// Host-side orchestration
// ---------------------------------------------------------------------------
extern "C" void kernel_launch(void* const* d_in, const int* in_sizes, int n_in,
                              void* d_out, int out_size, void* d_ws,
                              size_t ws_size, hipStream_t stream) {
  (void)in_sizes; (void)n_in; (void)ws_size; (void)out_size;
  constexpr int N0 = 100000, N1 = 50000, N2 = 25000;

  const float* x   = (const float*)d_in[0];
  const int*   ci0 = (const int*)d_in[1];
  const float* dp0 = (const float*)d_in[2];
  const int*   ci1 = (const int*)d_in[3];
  const float* dp1 = (const float*)d_in[4];
  const int*   pi1 = (const int*)d_in[5];
  const int*   ci2 = (const int*)d_in[6];
  const float* dp2 = (const float*)d_in[7];
  const int*   pi2 = (const int*)d_in[8];
  const float* w11 = (const float*)d_in[9];
  const float* b11 = (const float*)d_in[10];
  const float* w12 = (const float*)d_in[11];
  const float* b12 = (const float*)d_in[12];
  const float* w21 = (const float*)d_in[13];
  const float* b21 = (const float*)d_in[14];
  const float* w22 = (const float*)d_in[15];
  const float* b22 = (const float*)d_in[16];
  const float* w31 = (const float*)d_in[17];
  const float* b31 = (const float*)d_in[18];
  const float* wl  = (const float*)d_in[19];
  const float* bl  = (const float*)d_in[20];

  // carve workspace (256-byte aligned slices)
  char* ws = (char*)d_ws;
  size_t off = 0;
  auto carve = [&](size_t bytes) -> char* {
    off = (off + 255) & ~(size_t)255;
    char* p = ws + off;
    off += bytes;
    return p;
  };
  __bf16* xbf = (__bf16*)carve((size_t)N0 * 3 * 2);
  __bf16* h1  = (__bf16*)carve((size_t)N0 * 32 * 2);
  __bf16* h2  = (__bf16*)carve((size_t)N0 * 32 * 2);
  __bf16* p1  = (__bf16*)carve((size_t)N1 * 32 * 2);
  __bf16* h3  = (__bf16*)carve((size_t)N1 * 64 * 2);
  __bf16* h4  = (__bf16*)carve((size_t)N1 * 64 * 2);
  __bf16* p2  = (__bf16*)carve((size_t)N2 * 64 * 2);
  __bf16* h5  = (__bf16*)carve((size_t)N2 * 128 * 2);

  auto slabs = [](int csrc, int depth) {
    return (csrc * 9 + (depth ? 9 : 0) + 31) / 32;
  };
  auto fragElems = [&](int csrc, int depth, int cout) {
    return (size_t)slabs(csrc, depth) * (cout / 16) * 512;
  };
  __bf16* Bp0a = (__bf16*)carve(fragElems(3, 1, 32) * 2);
  __bf16* Bp0b = (__bf16*)carve(fragElems(32, 0, 32) * 2);
  __bf16* Bp1a = (__bf16*)carve(fragElems(32, 1, 64) * 2);
  __bf16* Bp1b = (__bf16*)carve(fragElems(64, 0, 64) * 2);
  __bf16* Bp2a = (__bf16*)carve(fragElems(64, 1, 128) * 2);

  // 1) convert x to bf16
  {
    int n = N0 * 3;
    cvt_f32_to_bf16<<<(n + 255) / 256, 256, 0, stream>>>(x, xbf, n);
  }
  // 2) repack all conv weights into WMMA B-fragment layout
  auto pack = [&](const float* w, __bf16* Bp, int csrc, int depth, int cout) {
    int n = (int)fragElems(csrc, depth, cout);
    pack_w_kernel<<<(n + 255) / 256, 256, 0, stream>>>(
        w, Bp, csrc, csrc + depth, cout, slabs(csrc, depth), depth);
  };
  pack(w11, Bp0a, 3, 1, 32);
  pack(w12, Bp0b, 32, 0, 32);
  pack(w21, Bp1a, 32, 1, 64);
  pack(w22, Bp1b, 64, 0, 64);
  pack(w31, Bp2a, 64, 1, 128);

  const int TB = WAVES_PER_BLOCK * 32;  // 128 threads, 64 points per block
  auto blocks = [](int n) { return (n + 63) / 64; };

  // level 0
  conv_kernel<3, true, 32, false>
      <<<blocks(N0), TB, 0, stream>>>(xbf, ci0, dp0, Bp0a, b11, h1, N0);
  conv_kernel<32, false, 32, true>
      <<<blocks(N0), TB, 0, stream>>>(h1, ci0, nullptr, Bp0b, b12, h2, N0);
  // level 1
  pool_kernel<32><<<(N1 + 255) / 256, 256, 0, stream>>>(h2, pi1, p1, N1);
  conv_kernel<32, true, 64, true>
      <<<blocks(N1), TB, 0, stream>>>(p1, ci1, dp1, Bp1a, b21, h3, N1);
  conv_kernel<64, false, 64, true>
      <<<blocks(N1), TB, 0, stream>>>(h3, ci1, nullptr, Bp1b, b22, h4, N1);
  // level 2
  pool_kernel<64><<<(N2 + 255) / 256, 256, 0, stream>>>(h4, pi2, p2, N2);
  conv_kernel<64, true, 128, true>
      <<<blocks(N2), TB, 0, stream>>>(p2, ci2, dp2, Bp2a, b31, h5, N2);
  // latent 1x1
  linear_kernel<<<(N2 + 255) / 256, 256, 0, stream>>>(h5, wl, bl,
                                                      (float*)d_out, N2);
}
